// Scaling_Attention_81801947119897
// MI455X (gfx1250) — matrix operand order repaired and checked by reading, not compiled, not run on hardware
//
#include <hip/hip_runtime.h>
#include <hip/hip_bf16.h>

// ---------------------------------------------------------------------------
// CDNA5 (gfx1250) fused kernel for the "Scaling Attention" block.
// All GEMMs on v_wmma_f32_16x16x32_f16 (wave32); fragments fed by b128 loads:
//   A: 2x global_load_b128 per fragment (contiguous k order, same as swizzled B)
//   B: LDS pre-swizzled into fragment order -> 2x ds_load_b128 per fragment
// ---------------------------------------------------------------------------

typedef __attribute__((ext_vector_type(16))) _Float16 v16h;
typedef __attribute__((ext_vector_type(8)))  _Float16 v8h;
typedef __attribute__((ext_vector_type(8)))  float    v8f;

#define EPS_BN 1e-5f

// ---- workspace layout (bytes) ---------------------------------------------
static constexpr size_t XN_OFF    = 0;
static constexpr size_t XN_BYTES  = (size_t)4 * 144 * 64 * 64 * 4;   // 9,437,184
static constexpr size_t WINH_OFF  = XN_OFF + XN_BYTES;
static constexpr size_t WINH_B    = (size_t)64 * 160 * 2;
static constexpr size_t W1H_OFF   = WINH_OFF + WINH_B;
static constexpr size_t W1H_B     = (size_t)64 * 576 * 2;
static constexpr size_t W2H_OFF   = W1H_OFF + W1H_B;
static constexpr size_t W2H_B     = (size_t)64 * 576 * 2;
static constexpr size_t WOUTH_OFF = W2H_OFF + W2H_B;
static constexpr size_t WOUTH_B   = (size_t)144 * 64 * 2;
static constexpr size_t BN_OFF    = WOUTH_OFF + WOUTH_B;
static constexpr size_t BN_B      = (size_t)672 * 4;
static constexpr size_t Z2G_OFF   = (BN_OFF + BN_B + 255) & ~(size_t)255;  // f16[16384][576]

// ---- fragment helpers ------------------------------------------------------
// A 16x32 (MxK), row-major source with leading dim `ld` (halfs).
// Lane(m = lane&15, half = lane>>4) needs
// K = k0 + half*16 + {0..15} (contiguous, matches swizzled B)  -> 2x b128.
__device__ __forceinline__ v16h load_A_frag(const _Float16* A, int ld, int m0, int k0, int lane) {
  int m = m0 + (lane & 15);
  int half = lane >> 4;
  const _Float16* row = A + (size_t)m * ld + k0 + (half << 4);
  union { v16h v; v8h h[2]; } u;
  u.h[0] = *(const v8h*)(row);
  u.h[1] = *(const v8h*)(row + 8);
  return u.v;
}

// B fragment element i (0..15) corresponds to K = k0 + half*16 + i, N = lane&15.
// Swizzled LDS layout: slot [(tile)*32 + lane] holds those 16 contiguous halfs.
__device__ __forceinline__ v16h load_B_sw(const _Float16* Bsw, int tile, int lane) {
  const _Float16* p = Bsw + (size_t)((tile << 5) + lane) * 16;
  union { v16h v; v8h h[2]; } u;
  u.h[0] = *(const v8h*)(p);
  u.h[1] = *(const v8h*)(p + 8);
  return u.v;
}

// scatter index for single-N-tile swizzled B: element (k, n), n in [0,16)
__device__ __forceinline__ int bswz(int k, int n) {
  return ((((k >> 5) << 1) + ((k >> 4) & 1)) * 16 + n) * 16 + (k & 15);
}
// scatter index for two-N-tile swizzled B: element (k, col), col in [0,32)
__device__ __forceinline__ int bswz2(int k, int col) {
  int tile = ((k >> 5) << 1) + (col >> 4);           // kstep*2 + nt
  int lane = (((k >> 4) & 1) << 4) + (col & 15);     // half*16 + n
  return ((tile << 5) + lane) * 16 + (k & 15);
}

__device__ __forceinline__ v8f wmma16(v16h a, v16h b, v8f c) {
  return __builtin_amdgcn_wmma_f32_16x16x32_f16(false, a, false, b, (short)0, c, false, false);
}

// ---------------------------------------------------------------------------
// prep: weight f16 conversion + BN folding
// ---------------------------------------------------------------------------
struct PrepArgs {
  const float *w_in, *g_in, *b_in, *m_in, *v_in;
  const float *w1, *g1, *b1, *m1, *v1;
  const float *w2, *g2, *b2, *m2, *v2;
  const float *w_out, *g_out, *b_out, *m_out, *v_out;
  _Float16 *winh, *w1h, *w2h, *wouth;
  float *bn;
};

__global__ __launch_bounds__(256) void prep_kernel(PrepArgs p) {
  int i0 = blockIdx.x * blockDim.x + threadIdx.x;
  int st = gridDim.x * blockDim.x;
  for (int i = i0; i < 64 * 160; i += st) {
    int o = i / 160, c = i - o * 160;
    p.winh[i] = (c < 144) ? (_Float16)p.w_in[o * 144 + c] : (_Float16)0.0f;
  }
  for (int i = i0; i < 64 * 576; i += st) p.w1h[i] = (_Float16)p.w1[i];   // OIHW flat == [64][576]
  for (int i = i0; i < 64 * 576; i += st) p.w2h[i] = (_Float16)p.w2[i];
  for (int i = i0; i < 144 * 64; i += st) p.wouth[i] = (_Float16)p.w_out[i];
  for (int i = i0; i < 64; i += st) {
    float s;
    s = p.g_in[i] * rsqrtf(p.v_in[i] + EPS_BN); p.bn[i]       = s; p.bn[64 + i]  = p.b_in[i] - p.m_in[i] * s;
    s = p.g1[i]   * rsqrtf(p.v1[i]   + EPS_BN); p.bn[128 + i] = s; p.bn[192 + i] = p.b1[i]   - p.m1[i]   * s;
    s = p.g2[i]   * rsqrtf(p.v2[i]   + EPS_BN); p.bn[256 + i] = s; p.bn[320 + i] = p.b2[i]   - p.m2[i]   * s;
  }
  for (int i = i0; i < 144; i += st) {
    float s = p.g_out[i] * rsqrtf(p.v_out[i] + EPS_BN);
    p.bn[384 + i] = s; p.bn[528 + i] = p.b_out[i] - p.m_out[i] * s;
  }
}

// ---------------------------------------------------------------------------
// normalize: per-pixel channel L2 norm (C=144)
// ---------------------------------------------------------------------------
__global__ __launch_bounds__(256) void normalize_kernel(const float* __restrict__ x,
                                                        float* __restrict__ xn) {
  int pix = blockIdx.x * 256 + threadIdx.x;          // 16384 pixels
  int b = pix >> 12, hw = pix & 4095;
  const float* xb = x + (size_t)b * 144 * 4096 + hw;
  float ss = 0.f;
  for (int c = 0; c < 144; ++c) { float v = xb[(size_t)c * 4096]; ss += v * v; }
  float d = sqrtf(ss); d = d > 1e-12f ? d : 1e-12f;
  float inv = 1.0f / d;
  float* ob = xn + (size_t)b * 144 * 4096 + hw;
  for (int c = 0; c < 144; ++c) ob[(size_t)c * 4096] = xb[(size_t)c * 4096] * inv;
}

// ---------------------------------------------------------------------------
// k2: per-pixel  y -> z1 (WMMA) -> im2col -> z2 (WMMA) -> global f16
// one workgroup (8 wave32) per pixel
// ---------------------------------------------------------------------------
__global__ __launch_bounds__(256) void k2_kernel(const float* __restrict__ xn,
                                                 const _Float16* __restrict__ winh,
                                                 const _Float16* __restrict__ w1h,
                                                 const float* __restrict__ bn,
                                                 _Float16* __restrict__ z2g) {
  __shared__ __align__(16) _Float16 BySw[5 * 2 * 32 * 16];   // y swizzled: 5 ksteps x 2 ntiles
  __shared__ __align__(16) _Float16 z1s[64 * 32];            // z1 plain: 64 x 25 (padded 32)
  __shared__ __align__(16) _Float16 z1colSw[18 * 32 * 16];   // im2col swizzled: 18 ksteps
  __shared__ float center[144];

  int tid = threadIdx.x, lane = tid & 31, wave = tid >> 5;
  int pix = blockIdx.x;
  int b = pix >> 12, hw = pix & 4095;
  int h = hw >> 6, w = hw & 63;
  const float* xb = xn + (size_t)b * 144 * 4096;

  for (int i = tid; i < 5 * 2 * 32 * 16; i += 256) BySw[i] = (_Float16)0.0f;
  if (tid < 144) center[tid] = xb[(size_t)tid * 4096 + hw];
  __syncthreads();

  // y[c][k] = xn[c, p+off_k] * xn[c, p]  (zero-padded 5x5 neighborhood), scattered swizzled
  for (int i = tid; i < 144 * 25; i += 256) {
    int c = i / 25, k = i - c * 25;
    int hh = h + k / 5 - 2, ww = w + k % 5 - 2;
    float v = 0.f;
    if (hh >= 0 && hh < 64 && ww >= 0 && ww < 64) v = xb[(size_t)c * 4096 + hh * 64 + ww];
    BySw[bswz2(c, k)] = (_Float16)(v * center[c]);
  }
  __syncthreads();

  // stage B: z1[64][25] = W_in[64][144] @ y   (4 M-tiles x 2 N-tiles over 8 waves)
  {
    int mt = wave >> 1, nt = wave & 1;
    v8f acc = {};
#pragma unroll
    for (int ks = 0; ks < 5; ++ks) {
      v16h a  = load_A_frag(winh, 160, mt * 16, ks * 32, lane);
      v16h bf = load_B_sw(BySw, ks * 2 + nt, lane);
      acc = wmma16(a, bf, acc);
    }
    int half = lane >> 4, n = lane & 15, col = nt * 16 + n;
#pragma unroll
    for (int r = 0; r < 8; ++r) {
      int o = mt * 16 + half * 8 + r;
      float z = acc[r] * bn[o] + bn[64 + o];
      z = z > 0.f ? z : 0.f;
      if (col < 25) z1s[o * 32 + col] = (_Float16)z;
    }
  }
  __syncthreads();

  for (int i = tid; i < 18 * 32 * 16; i += 256) z1colSw[i] = (_Float16)0.0f;
  __syncthreads();
  // im2col: col[c*9 + ry*3+rx][qy*3+qx] = z1[c][(qy+ry)*5 + qx+rx], scattered swizzled
  for (int i = tid; i < 576 * 9; i += 256) {
    int kk = i / 9, q = i - kk * 9;
    int c = kk / 9, r = kk - c * 9;
    int ry = r / 3, rx = r - ry * 3, qy = q / 3, qx = q - qy * 3;
    z1colSw[bswz(kk, q)] = z1s[c * 32 + (qy + ry) * 5 + (qx + rx)];
  }
  __syncthreads();

  // stage C: z2[64][9] = W1col[64][576] @ z1col  (4 M-tiles, waves 0..3, 18 K-steps)
  if (wave < 4) {
    int mt = wave;
    v8f acc = {};
#pragma unroll 6
    for (int ks = 0; ks < 18; ++ks) {
      v16h a  = load_A_frag(w1h, 576, mt * 16, ks * 32, lane);
      v16h bf = load_B_sw(z1colSw, ks, lane);
      acc = wmma16(a, bf, acc);
    }
    int half = lane >> 4, n = lane & 15;
#pragma unroll
    for (int r = 0; r < 8; ++r) {
      int o = mt * 16 + half * 8 + r;
      float z = acc[r] * bn[128 + o] + bn[192 + o];
      z = z > 0.f ? z : 0.f;
      if (n < 9) z2g[(size_t)pix * 576 + o * 9 + n] = (_Float16)z;
    }
  }
}

// ---------------------------------------------------------------------------
// k3: 16 pixels per workgroup:  z3[64][16] (WMMA) -> out[144][16] (WMMA) -> BN
// ---------------------------------------------------------------------------
__global__ __launch_bounds__(256) void k3_kernel(const _Float16* __restrict__ z2g,
                                                 const _Float16* __restrict__ w2h,
                                                 const _Float16* __restrict__ wouth,
                                                 const float* __restrict__ bn,
                                                 float* __restrict__ out) {
  __shared__ __align__(16) _Float16 B2Sw[18 * 32 * 16];   // z2col swizzled: K=576 x N=16 px
  __shared__ __align__(16) _Float16 z3Sw[2 * 32 * 16];    // z3 swizzled:    K=64  x N=16 px

  int tid = threadIdx.x, lane = tid & 31, wave = tid >> 5;
  int pix0 = blockIdx.x * 16;

  for (int i = tid; i < 576 * 16; i += 256) {
    int kk = i >> 4, n = i & 15;
    B2Sw[bswz(kk, n)] = z2g[(size_t)(pix0 + n) * 576 + kk];
  }
  __syncthreads();

  // stage D: z3[64][16] = W2col[64][576] @ z2col  (waves 0..3, 18 K-steps)
  if (wave < 4) {
    int mt = wave;
    v8f acc = {};
#pragma unroll 6
    for (int ks = 0; ks < 18; ++ks) {
      v16h a  = load_A_frag(w2h, 576, mt * 16, ks * 32, lane);
      v16h bf = load_B_sw(B2Sw, ks, lane);
      acc = wmma16(a, bf, acc);
    }
    int half = lane >> 4, n = lane & 15;
#pragma unroll
    for (int r = 0; r < 8; ++r) {
      int o = mt * 16 + half * 8 + r;
      float z = acc[r] * bn[256 + o] + bn[320 + o];
      z = z > 0.f ? z : 0.f;
      z3Sw[bswz(o, n)] = (_Float16)z;
    }
  }
  __syncthreads();

  // stage E: out[144][16] = W_out[144][64] @ z3  (9 M-tiles over 8 waves, 2 K-steps)
  int bb = pix0 >> 12, hw0 = pix0 & 4095;
  for (int mt = wave; mt < 9; mt += 8) {
    v8f acc = {};
#pragma unroll
    for (int ks = 0; ks < 2; ++ks) {
      v16h a  = load_A_frag(wouth, 64, mt * 16, ks * 32, lane);
      v16h bf = load_B_sw(z3Sw, ks, lane);
      acc = wmma16(a, bf, acc);
    }
    int half = lane >> 4, n = lane & 15;
#pragma unroll
    for (int r = 0; r < 8; ++r) {
      int oc = mt * 16 + half * 8 + r;
      float z = acc[r] * bn[384 + oc] + bn[528 + oc];
      out[(size_t)bb * 144 * 4096 + (size_t)oc * 4096 + hw0 + n] = z;
    }
  }
}

// ---------------------------------------------------------------------------
extern "C" void kernel_launch(void* const* d_in, const int* in_sizes, int n_in,
                              void* d_out, int out_size, void* d_ws, size_t ws_size,
                              hipStream_t stream) {
  (void)in_sizes; (void)n_in; (void)out_size; (void)ws_size;
  const float* x = (const float*)d_in[0];
  PrepArgs p;
  p.w_in  = (const float*)d_in[1];
  p.g_in  = (const float*)d_in[2];  p.b_in = (const float*)d_in[3];
  p.m_in  = (const float*)d_in[4];  p.v_in = (const float*)d_in[5];
  p.w1    = (const float*)d_in[6];
  p.g1    = (const float*)d_in[7];  p.b1   = (const float*)d_in[8];
  p.m1    = (const float*)d_in[9];  p.v1   = (const float*)d_in[10];
  p.w2    = (const float*)d_in[11];
  p.g2    = (const float*)d_in[12]; p.b2   = (const float*)d_in[13];
  p.m2    = (const float*)d_in[14]; p.v2   = (const float*)d_in[15];
  p.w_out = (const float*)d_in[16];
  p.g_out = (const float*)d_in[17]; p.b_out = (const float*)d_in[18];
  p.m_out = (const float*)d_in[19]; p.v_out = (const float*)d_in[20];

  char* ws = (char*)d_ws;
  float*    xn    = (float*)(ws + XN_OFF);
  _Float16* winh  = (_Float16*)(ws + WINH_OFF);
  _Float16* w1h   = (_Float16*)(ws + W1H_OFF);
  _Float16* w2h   = (_Float16*)(ws + W2H_OFF);
  _Float16* wouth = (_Float16*)(ws + WOUTH_OFF);
  float*    bn    = (float*)(ws + BN_OFF);
  _Float16* z2g   = (_Float16*)(ws + Z2G_OFF);
  p.winh = winh; p.w1h = w1h; p.w2h = w2h; p.wouth = wouth; p.bn = bn;

  float* out = (float*)d_out;

  prep_kernel<<<64, 256, 0, stream>>>(p);
  normalize_kernel<<<64, 256, 0, stream>>>(x, xn);
  k2_kernel<<<16384, 256, 0, stream>>>(xn, winh, w1h, bn, z2g);
  k3_kernel<<<1024, 256, 0, stream>>>(z2g, w2h, wouth, bn, out);
}
